// PointPatchEmbed_42219528520350
// MI455X (gfx1250) — compile-verified, matrix operand order checked by reading, not run on hardware
//
#include <hip/hip_runtime.h>
#include <hip/hip_bf16.h>
#include <math.h>

typedef __attribute__((ext_vector_type(16))) _Float16 v16h;
typedef __attribute__((ext_vector_type(8)))  float    v8f;

#define NPTS   8192
#define NBATCH 32
#define NGRP   512
#define KNB    32
#define HID    128
#define EMB    384
#define SH_STRIDE 136   // padded f16 row stride for hidden activations in LDS

// ---------------------------------------------------------------------------
// Wave (32-lane) argmax / argmin with first-index tie-break
// ---------------------------------------------------------------------------
__device__ __forceinline__ void wave_argmax(float& v, int& i) {
#pragma unroll
    for (int off = 16; off > 0; off >>= 1) {
        float ov = __shfl_xor(v, off, 32);
        int   oi = __shfl_xor(i, off, 32);
        if (ov > v || (ov == v && oi < i)) { v = ov; i = oi; }
    }
}
__device__ __forceinline__ void wave_argmin(float& v, int& i) {
#pragma unroll
    for (int off = 16; off > 0; off >>= 1) {
        float ov = __shfl_xor(v, off, 32);
        int   oi = __shfl_xor(i, off, 32);
        if (ov < v || (ov == v && oi < i)) { v = ov; i = oi; }
    }
}

// ---------------------------------------------------------------------------
// Kernel 0: transpose W2 (128x384 f32, row-major) -> W2t (384x128 f16)
// so WMMA B fragments become two contiguous 16B loads per lane.
// ---------------------------------------------------------------------------
__global__ void prep_w2_kernel(const float* __restrict__ W2, _Float16* __restrict__ W2t) {
    int i = blockIdx.x * 256 + threadIdx.x;           // i over 384*128
    if (i < EMB * HID) {
        int n = i >> 7;           // 0..383
        int k = i & 127;          // 0..127
        W2t[i] = (_Float16)W2[k * EMB + n];
    }
}

// ---------------------------------------------------------------------------
// Kernel 1: farthest point sampling. One block (1024 thr = 32 waves) per batch.
// Per-point running min-distance lives in registers (8 points / thread).
// Writes centers_xyz (B,G,3) directly to the output buffer.
// ---------------------------------------------------------------------------
__global__ __launch_bounds__(1024) void fps_kernel(const float* __restrict__ xyz,
                                                   float* __restrict__ centers_out) {
    const int b = blockIdx.x;
    const int t = threadIdx.x;
    const int w = t >> 5, lane = t & 31;
    const float* xb = xyz + (size_t)b * NPTS * 3;

    __shared__ float s_acc[3];
    __shared__ float s_c[3];
    __shared__ int   s_fi;
    __shared__ float r_v[32];
    __shared__ int   r_i[32];

    float px[8], py[8], pz[8], dmin[8];
    float sx = 0.f, sy = 0.f, sz = 0.f;
#pragma unroll
    for (int j = 0; j < 8; ++j) {
        int idx = t + j * 1024;
        px[j] = xb[idx * 3 + 0];
        py[j] = xb[idx * 3 + 1];
        pz[j] = xb[idx * 3 + 2];
        sx += px[j]; sy += py[j]; sz += pz[j];
        dmin[j] = 1e10f;
    }

    if (t < 3) s_acc[t] = 0.f;
    __syncthreads();
#pragma unroll
    for (int off = 16; off > 0; off >>= 1) {
        sx += __shfl_xor(sx, off, 32);
        sy += __shfl_xor(sy, off, 32);
        sz += __shfl_xor(sz, off, 32);
    }
    if (lane == 0) {
        atomicAdd(&s_acc[0], sx);
        atomicAdd(&s_acc[1], sy);
        atomicAdd(&s_acc[2], sz);
    }
    __syncthreads();
    const float inv = 1.0f / (float)NPTS;
    float mx = s_acc[0] * inv, my = s_acc[1] * inv, mz = s_acc[2] * inv;

    // initial farthest = argmax dist-to-mean
    float bv = -1.f; int bi = 0;
#pragma unroll
    for (int j = 0; j < 8; ++j) {
        int idx = t + j * 1024;
        float dx = px[j] - mx, dy = py[j] - my, dz = pz[j] - mz;
        float d = dx * dx + dy * dy + dz * dz;
        if (d > bv) { bv = d; bi = idx; }
    }
    wave_argmax(bv, bi);
    if (lane == 0) { r_v[w] = bv; r_i[w] = bi; }
    __syncthreads();
    if (t == 0) {
        float fv = r_v[0]; int fi = r_i[0];
        for (int k = 1; k < 32; ++k)
            if (r_v[k] > fv || (r_v[k] == fv && r_i[k] < fi)) { fv = r_v[k]; fi = r_i[k]; }
        s_fi = fi;
    }
    __syncthreads();

    for (int g = 0; g < NGRP; ++g) {
        int fi = s_fi;
        if ((fi & 1023) == t) {       // owner publishes centroid coords
            int j = fi >> 10;
            s_c[0] = px[j]; s_c[1] = py[j]; s_c[2] = pz[j];
        }
        __syncthreads();
        float cx = s_c[0], cy = s_c[1], cz = s_c[2];
        if (t == 0) {
            size_t o = ((size_t)b * NGRP + g) * 3;
            centers_out[o + 0] = cx; centers_out[o + 1] = cy; centers_out[o + 2] = cz;
        }
        float lv = -1.f; int li = 0;
#pragma unroll
        for (int j = 0; j < 8; ++j) {
            int idx = t + j * 1024;
            float dx = px[j] - cx, dy = py[j] - cy, dz = pz[j] - cz;
            float d = dx * dx + dy * dy + dz * dz;
            dmin[j] = fminf(dmin[j], d);
            if (dmin[j] > lv) { lv = dmin[j]; li = idx; }
        }
        wave_argmax(lv, li);
        if (lane == 0) { r_v[w] = lv; r_i[w] = li; }
        __syncthreads();
        if (t == 0) {
            float fv = r_v[0]; int fj = r_i[0];
            for (int k = 1; k < 32; ++k)
                if (r_v[k] > fv || (r_v[k] == fv && r_i[k] < fj)) { fv = r_v[k]; fj = r_i[k]; }
            s_fi = fj;
        }
        __syncthreads();
    }
}

// ---------------------------------------------------------------------------
// Kernel 2 (fused): per (b,g) group — kNN top-32, gather, layer1 (f32 VALU +
// exact GELU), layer2 via v_wmma_f32_16x16x32_f16, max over K, +b2.
// One block (256 thr = 8 waves) per group.
// ---------------------------------------------------------------------------
__global__ __launch_bounds__(256) void group_mlp_kernel(
    const float* __restrict__ xyz,
    const float* __restrict__ W1, const float* __restrict__ b1,
    const _Float16* __restrict__ W2t, const float* __restrict__ b2,
    const float* __restrict__ centers,
    float* __restrict__ tokens, int* __restrict__ gidx_out) {

    __shared__ float    sdist[NPTS];              // 32 KB
    __shared__ _Float16 sh[KNB * SH_STRIDE];      // hidden activations, f16, padded
    __shared__ float    srel[KNB][3];
    __shared__ int      sgi[KNB];
    __shared__ float    r_v[8];
    __shared__ int      r_i[8];

    const int bg  = blockIdx.x;                   // 0 .. B*G-1
    const int b   = bg >> 9;                      // /512
    const int t   = threadIdx.x;
    const int w   = t >> 5, lane = t & 31;
    const float* xb = xyz + (size_t)b * NPTS * 3;

    const float cx = centers[(size_t)bg * 3 + 0];
    const float cy = centers[(size_t)bg * 3 + 1];
    const float cz = centers[(size_t)bg * 3 + 2];

    // ---- squared distances to center, into LDS ----
#pragma unroll
    for (int k = 0; k < NPTS / 256; ++k) {
        int idx = t + k * 256;
        float dx = xb[idx * 3 + 0] - cx;
        float dy = xb[idx * 3 + 1] - cy;
        float dz = xb[idx * 3 + 2] - cz;
        sdist[idx] = dx * dx + dy * dy + dz * dz;
    }
    __syncthreads();

    // ---- 32 rounds of block-wide argmin with marking ----
    for (int r = 0; r < KNB; ++r) {
        float lv = 3.0e38f; int li = 0;
#pragma unroll
        for (int k = 0; k < NPTS / 256; ++k) {
            int idx = t + k * 256;                // stride-256: conflict-free banks
            float v = sdist[idx];
            if (v < lv) { lv = v; li = idx; }
        }
        wave_argmin(lv, li);
        if (lane == 0) { r_v[w] = lv; r_i[w] = li; }
        __syncthreads();
        if (t == 0) {
            float fv = r_v[0]; int fi = r_i[0];
            for (int k = 1; k < 8; ++k)
                if (r_v[k] < fv || (r_v[k] == fv && r_i[k] < fi)) { fv = r_v[k]; fi = r_i[k]; }
            sgi[r] = fi;
            sdist[fi] = 3.0e38f;                  // mark taken
            gidx_out[(size_t)bg * KNB + r] = fi;
        }
        __syncthreads();
    }

    // ---- gather relative coordinates ----
    if (t < KNB) {
        int id = sgi[t];
        srel[t][0] = xb[id * 3 + 0] - cx;
        srel[t][1] = xb[id * 3 + 1] - cy;
        srel[t][2] = xb[id * 3 + 2] - cz;
    }
    __syncthreads();

    // ---- layer1: h = gelu(rel @ W1 + b1), f32 VALU, store f16 to LDS ----
    for (int i = t; i < KNB * HID; i += 256) {
        int row = i >> 7, col = i & 127;
        float x = srel[row][0] * W1[col]
                + srel[row][1] * W1[HID + col]
                + srel[row][2] * W1[2 * HID + col]
                + b1[col];
        float ge = 0.5f * x * (1.0f + erff(x * 0.70710678118654752f));
        sh[row * SH_STRIDE + col] = (_Float16)ge;
    }
    __syncthreads();

    // ---- layer2: (32x128) @ (128x384) on WMMA, then max over the 32 rows ----
    const int lcol  = lane & 15;
    const int lhalf = lane >> 4;
#pragma unroll
    for (int i = 0; i < 3; ++i) {                 // 24 N-tiles / 8 waves
        const int nt = w + 8 * i;
        const int n0 = nt * 16;
        float colmax = -3.0e38f;
#pragma unroll
        for (int mt = 0; mt < 2; ++mt) {          // 32 rows = 2 M-tiles
            v8f acc = {};
            const int arow = mt * 16 + lcol;
#pragma unroll
            for (int kk = 0; kk < 4; ++kk) {      // K=128 = 4 x 32
                const int kb = kk * 32 + lhalf * 8;
                union { v16h v; _Float16 h[16]; } A, Bf;
                const _Float16* ap = &sh[arow * SH_STRIDE + kb];
                const _Float16* bp = W2t + (size_t)(n0 + lcol) * HID + kb;
#pragma unroll
                for (int j = 0; j < 8; ++j) {
                    A.h[j]      = ap[j];
                    A.h[8 + j]  = ap[16 + j];
                    Bf.h[j]     = bp[j];
                    Bf.h[8 + j] = bp[16 + j];
                }
                acc = __builtin_amdgcn_wmma_f32_16x16x32_f16(
                    false, A.v, false, Bf.v, (short)0, acc, false, false);
            }
            // lane holds rows (mt*16 + lhalf*8 + r) of column lcol
            float m = acc[0];
#pragma unroll
            for (int r = 1; r < 8; ++r) m = fmaxf(m, acc[r]);
            m = fmaxf(m, __shfl_xor(m, 16, 32));  // combine both row-halves
            colmax = fmaxf(colmax, m);
        }
        if (lane < 16)
            tokens[(size_t)bg * EMB + n0 + lane] = colmax + b2[n0 + lane];
    }
}

// ---------------------------------------------------------------------------
// Launch
// ---------------------------------------------------------------------------
extern "C" void kernel_launch(void* const* d_in, const int* in_sizes, int n_in,
                              void* d_out, int out_size, void* d_ws, size_t ws_size,
                              hipStream_t stream) {
    const float* xyz = (const float*)d_in[0];   // (32, 8192, 3)
    const float* W1  = (const float*)d_in[1];   // (3, 128)
    const float* b1  = (const float*)d_in[2];   // (128,)
    const float* W2  = (const float*)d_in[3];   // (128, 384)
    const float* b2  = (const float*)d_in[4];   // (384,)

    float* out      = (float*)d_out;
    float* tokens   = out;                                          // B*G*384
    float* centers  = out + (size_t)NBATCH * NGRP * EMB;            // B*G*3
    int*   gidx_out = (int*)(centers + (size_t)NBATCH * NGRP * 3);  // B*G*32

    _Float16* W2t = (_Float16*)d_ws;                                // 96 KB scratch

    prep_w2_kernel<<<(EMB * HID + 255) / 256, 256, 0, stream>>>(W2, W2t);
    fps_kernel<<<NBATCH, 1024, 0, stream>>>(xyz, centers);
    group_mlp_kernel<<<NBATCH * NGRP, 256, 0, stream>>>(
        xyz, W1, b1, W2t, b2, centers, tokens, gidx_out);
}
